// Stem_slic_V2_23845658427410
// MI455X (gfx1250) — matrix operand order repaired and checked by reading, not compiled, not run on hardware
//
#include <hip/hip_runtime.h>

typedef _Float16 v8h  __attribute__((ext_vector_type(8)));
typedef _Float16 v16h __attribute__((ext_vector_type(16)));
typedef float    v8f  __attribute__((ext_vector_type(8)));

#define NBATCH 32
#define NSEG   196
#define NCELL  196

// ---------------------------------------------------------------------------
// BN coefficient prep: scale = gamma/sqrt(var+eps), shift = (b-mean)*scale+beta
// ---------------------------------------------------------------------------
__global__ void bn_prep_kernel(const float* __restrict__ b, const float* __restrict__ g,
                               const float* __restrict__ be, const float* __restrict__ mn,
                               const float* __restrict__ vr, float* __restrict__ sA,
                               float* __restrict__ sB, int C) {
    int c = blockIdx.x * 64 + threadIdx.x;
    if (c >= C) return;
    float s = g[c] * rsqrtf(vr[c] + 1e-5f);
    sA[c] = s;
    sB[c] = (b[c] - mn[c]) * s + be[c];
}

// f32 weights -> f16 (layout [Cout][Cin*9] is already GEMM A layout)
__global__ void w2h_kernel(const float* __restrict__ w, _Float16* __restrict__ wh, int n) {
    int i = blockIdx.x * 256 + threadIdx.x;
    if (i < n) wh[i] = (_Float16)w[i];
}

// ---------------------------------------------------------------------------
// Layer 1: direct conv 3->96, stride 2, pad 1, 64x64 -> 32x32, BN+ReLU
// ---------------------------------------------------------------------------
__global__ void conv1_kernel(const float* __restrict__ x, const float* __restrict__ w,
                             const float* __restrict__ sA, const float* __restrict__ sB,
                             float* __restrict__ out) {
    int idx = blockIdx.x * 256 + threadIdx.x;          // ((b*96+co)*32+oy)*32+ox
    int ox = idx & 31; int t = idx >> 5;
    int oy = t & 31;   t >>= 5;
    int co = t % 96;   int b = t / 96;
    float acc = 0.f;
    #pragma unroll
    for (int ci = 0; ci < 3; ++ci)
        #pragma unroll
        for (int ky = 0; ky < 3; ++ky) {
            int iy = oy * 2 - 1 + ky;
            if ((unsigned)iy >= 64u) continue;
            #pragma unroll
            for (int kx = 0; kx < 3; ++kx) {
                int ix = ox * 2 - 1 + kx;
                if ((unsigned)ix >= 64u) continue;
                acc += x[((b * 3 + ci) * 64 + iy) * 64 + ix] *
                       w[((co * 3 + ci) * 3 + ky) * 3 + kx];
            }
        }
    out[idx] = fmaxf(acc * sA[co] + sB[co], 0.f);
}

// ---------------------------------------------------------------------------
// Layers 2-5: implicit GEMM via WMMA f16 (16x16x32, f32 accum).
// M = Cout, N = B*OH*OW, K = Cin*9 (all multiples of 32; M,N multiples of 64).
// Block = 256 threads = 8 waves; 64x64 output tile.
// Wave (wm2, wn4) with wm2 = wave>>2, wn4 = wave&3 owns rows [wm2*32,+32),
// cols [wn4*16,+16): 2 WMMAs per K-step sharing one B fragment.
// Software-pipelined: global fetch of step k0+32 overlaps WMMA of step k0.
// ---------------------------------------------------------------------------
__global__ __launch_bounds__(256) void conv_wmma_kernel(
    const float* __restrict__ in, const _Float16* __restrict__ wh,
    const float* __restrict__ sA, const float* __restrict__ sB,
    const float* __restrict__ pos, float* __restrict__ out,
    int Cin, int H, int W, int Cout, int OH, int OW,
    int stride, int pad, int relu) {
    const int K   = Cin * 9;
    const int OHW = OH * OW;

    __shared__ _Float16 Ah[64][40];   // [m][k], +8 pad halfs per row
    __shared__ _Float16 Bt[64][40];   // [n][k] (im2col, transposed)

    const int t    = threadIdx.x;
    const int lane = t & 31;
    const int wave = t >> 5;
    const int wm2  = wave >> 2;       // 0..1 : 32-row slab
    const int wn4  = wave & 3;        // 0..3 : 16-col slab
    const int r16  = lane & 15;
    const int hh   = lane >> 4;       // lane half

    const int n0 = blockIdx.x * 64;
    const int m0 = blockIdx.y * 64;

    // ---- hoist all loop-invariant staging indices (8 elements per thread) --
    int mA[8], nJ[8], kkA[8];         // tile-local coords
    int aOff[8];                      // weight element offset (add k0)
    int bBase[8];                     // image base offset  b*Cin*H*W
    int iy0[8], ix0[8];
    #pragma unroll
    for (int j = 0; j < 8; ++j) {
        int i  = t + 256 * j;         // 0..2047
        int mn = i >> 5;              // 0..63
        int kk = i & 31;
        mA[j] = mn; nJ[j] = mn; kkA[j] = kk;
        aOff[j] = (m0 + mn) * K + kk;
        int ng  = n0 + mn;
        int b   = ng / OHW;
        int rem = ng - b * OHW;
        int oy  = rem / OW;
        int ox  = rem - oy * OW;
        bBase[j] = b * Cin * H * W;
        iy0[j] = oy * stride - pad;
        ix0[j] = ox * stride - pad;
    }

    // ---- fragment LDS pointers (loop-invariant) ----------------------------
    const v8h* aP0 = (const v8h*)&Ah[wm2 * 32 + r16][8 * hh];        // subtile 0
    const v8h* aP1 = (const v8h*)&Ah[wm2 * 32 + r16][16 + 8 * hh];
    const v8h* aP2 = (const v8h*)&Ah[wm2 * 32 + 16 + r16][8 * hh];   // subtile 1
    const v8h* aP3 = (const v8h*)&Ah[wm2 * 32 + 16 + r16][16 + 8 * hh];
    const v8h* bP0 = (const v8h*)&Bt[wn4 * 16 + r16][8 * hh];
    const v8h* bP1 = (const v8h*)&Bt[wn4 * 16 + r16][16 + 8 * hh];

    v8f acc0 = {0.f, 0.f, 0.f, 0.f, 0.f, 0.f, 0.f, 0.f};
    v8f acc1 = {0.f, 0.f, 0.f, 0.f, 0.f, 0.f, 0.f, 0.f};

    _Float16 aReg[8];
    float    bReg[8];

    // fetch for a given k0 into registers
    auto fetch = [&](int k0) {
        #pragma unroll
        for (int j = 0; j < 8; ++j)
            aReg[j] = wh[aOff[j] + k0];
        #pragma unroll
        for (int j = 0; j < 8; ++j) {
            int k  = k0 + kkA[j];
            int ci = k / 9;            // constant-divisor: strength-reduced
            int r9 = k - ci * 9;
            int ky = r9 / 3;
            int kx = r9 - ky * 3;
            int iy = iy0[j] + ky;
            int ix = ix0[j] + kx;
            float v = 0.f;
            if ((unsigned)iy < (unsigned)H && (unsigned)ix < (unsigned)W)
                v = in[bBase[j] + (ci * H + iy) * W + ix];
            bReg[j] = v;
        }
    };

    fetch(0);
    for (int k0 = 0; k0 < K; k0 += 32) {
        __syncthreads();
        #pragma unroll
        for (int j = 0; j < 8; ++j) Ah[mA[j]][kkA[j]] = aReg[j];
        #pragma unroll
        for (int j = 0; j < 8; ++j) Bt[nJ[j]][kkA[j]] = (_Float16)bReg[j];
        __syncthreads();

        if (k0 + 32 < K) fetch(k0 + 32);   // overlap next fetch with WMMA

        v8h a0 = *aP0, a1 = *aP1, a2 = *aP2, a3 = *aP3;
        v8h b0 = *bP0, b1 = *bP1;
        v16h av0 = __builtin_shufflevector(a0, a1, 0,1,2,3,4,5,6,7,8,9,10,11,12,13,14,15);
        v16h av1 = __builtin_shufflevector(a2, a3, 0,1,2,3,4,5,6,7,8,9,10,11,12,13,14,15);
        v16h bv  = __builtin_shufflevector(b0, b1, 0,1,2,3,4,5,6,7,8,9,10,11,12,13,14,15);
        acc0 = __builtin_amdgcn_wmma_f32_16x16x32_f16(false, av0, false, bv,
                                                      (short)0, acc0, false, false);
        acc1 = __builtin_amdgcn_wmma_f32_16x16x32_f16(false, av1, false, bv,
                                                      (short)0, acc1, false, false);
    }

    // epilogue: D layout -> lane gives N, VGPR r gives M = r + 8*hh
    const int ng  = n0 + wn4 * 16 + r16;
    const int b   = ng / OHW;
    const int rem = ng - b * OHW;
    #pragma unroll
    for (int r = 0; r < 8; ++r) {
        int co0 = m0 + wm2 * 32 + 8 * hh + r;
        int co1 = co0 + 16;
        float y0 = acc0[r] * sA[co0] + sB[co0];
        float y1 = acc1[r] * sA[co1] + sB[co1];
        if (relu) { y0 = fmaxf(y0, 0.f); y1 = fmaxf(y1, 0.f); }
        if (pos)  { y0 += pos[co0 * OHW + rem]; y1 += pos[co1 * OHW + rem]; }
        out[(b * Cout + co0) * OHW + rem] = y0;
        out[(b * Cout + co1) * OHW + rem] = y1;
    }
}

// ---------------------------------------------------------------------------
// Segment mask: bilinear-resample one-hot masks 64->56 analytically, threshold
// >0.1, bump cnt[b][cell][seg] (cell = 4x4 window of the 14x14 grid).
// ---------------------------------------------------------------------------
__global__ void mask_kernel(const int* __restrict__ seg, int* __restrict__ cnt) {
    int idx = blockIdx.x * 256 + threadIdx.x;          // B*56*56 = 100352
    int b = idx / 3136;
    int r = idx - b * 3136;
    int y = r / 56, x = r - (r / 56) * 56;
    const float sc = 64.f / 56.f;
    float sy = (y + 0.5f) * sc - 0.5f;
    float sx = (x + 0.5f) * sc - 0.5f;
    int y0 = (int)floorf(sy), x0 = (int)floorf(sx);
    float fy = sy - y0, fx = sx - x0;
    int y0c = min(max(y0, 0), 63), y1c = min(max(y0 + 1, 0), 63);
    int x0c = min(max(x0, 0), 63), x1c = min(max(x0 + 1, 0), 63);
    const int* sb = seg + b * 4096;
    int s[4]   = { sb[y0c * 64 + x0c], sb[y0c * 64 + x1c],
                   sb[y1c * 64 + x0c], sb[y1c * 64 + x1c] };
    float wt[4] = { (1.f - fy) * (1.f - fx), (1.f - fy) * fx,
                    fy * (1.f - fx),         fy * fx };
    int cell = (y >> 2) * 14 + (x >> 2);
    #pragma unroll
    for (int i = 0; i < 4; ++i) {
        bool dup = false;
        for (int j = 0; j < i; ++j) dup = dup || (s[j] == s[i]);
        if (dup) continue;
        float wsum = 0.f;
        for (int j = i; j < 4; ++j) if (s[j] == s[i]) wsum += wt[j];
        if (wsum > 0.1f)
            atomicAdd(&cnt[(b * NCELL + cell) * NSEG + s[i]], 1);
    }
}

__global__ void den_kernel(const int* __restrict__ cnt, float* __restrict__ den) {
    int idx = blockIdx.x * 256 + threadIdx.x;          // B*196
    if (idx >= NBATCH * NSEG) return;
    int b = idx / NSEG, m = idx - (idx / NSEG) * NSEG;
    int s = 0;
    for (int c = 0; c < NCELL; ++c) s += cnt[(b * NCELL + c) * NSEG + m];
    den[idx] = (float)s;
}

// x_out[b,f,m] = sum_cell feat[b,f,cell]*cnt[b,cell,m] / den[b,m]
// grid: (768/16, 32); block 256; 16 feature rows staged in LDS.
__global__ __launch_bounds__(256) void feat_kernel(const float* __restrict__ feat,
                                                   const int* __restrict__ cnt,
                                                   const float* __restrict__ den,
                                                   float* __restrict__ xout) {
    const int b  = blockIdx.y;
    const int f0 = blockIdx.x * 16;
    __shared__ float fsh[16][NCELL];
    for (int i = threadIdx.x; i < 16 * NCELL; i += 256) {
        int f = i / NCELL, c = i - (i / NCELL) * NCELL;
        fsh[f][c] = feat[(b * 768 + f0 + f) * NCELL + c];
    }
    __syncthreads();
    int m = threadIdx.x;
    if (m >= NSEG) return;
    float d   = den[b * NSEG + m];
    float inv = d > 0.f ? 1.f / d : 0.f;
    float acc[16];
    #pragma unroll
    for (int f = 0; f < 16; ++f) acc[f] = 0.f;
    const int* cp = cnt + b * NCELL * NSEG + m;
    for (int c = 0; c < NCELL; ++c) {
        float cv = (float)cp[c * NSEG];
        #pragma unroll
        for (int f = 0; f < 16; ++f) acc[f] += fsh[f][c] * cv;
    }
    #pragma unroll
    for (int f = 0; f < 16; ++f)
        xout[(b * 768 + f0 + f) * NSEG + m] = acc[f] * inv;
}

__global__ void segcopy_kernel(const int* __restrict__ seg, float* __restrict__ out, int n) {
    int i = blockIdx.x * 256 + threadIdx.x;
    if (i < n) out[i] = (float)seg[i];
}

// ---------------------------------------------------------------------------
extern "C" void kernel_launch(void* const* d_in, const int* in_sizes, int n_in,
                              void* d_out, int out_size, void* d_ws, size_t ws_size,
                              hipStream_t stream) {
    (void)in_sizes; (void)n_in; (void)out_size; (void)ws_size;

    const float* x   = (const float*)d_in[0];
    const int*   seg = (const int*)d_in[1];
    const float *W[5], *Bb[5], *G[5], *Be[5], *Mn[5], *Vr[5];
    for (int l = 0; l < 5; ++l) {
        W[l]  = (const float*)d_in[2 + 6 * l + 0];
        Bb[l] = (const float*)d_in[2 + 6 * l + 1];
        G[l]  = (const float*)d_in[2 + 6 * l + 2];
        Be[l] = (const float*)d_in[2 + 6 * l + 3];
        Mn[l] = (const float*)d_in[2 + 6 * l + 4];
        Vr[l] = (const float*)d_in[2 + 6 * l + 5];
    }
    const float* pos = (const float*)d_in[32];

    float* dout  = (float*)d_out;
    float* xout  = dout + NBATCH * 64 * 64;                       // x_out
    float* cfeat = xout + NBATCH * 768 * NSEG;                    // x_convFeat

    // workspace layout
    char* wsb = (char*)d_ws;
    size_t off = 0;
    auto take = [&](size_t bytes) -> void* {
        void* p = wsb + off;
        off = (off + bytes + 255) & ~(size_t)255;
        return p;
    };
    float*    act1 = (float*)take((size_t)NBATCH * 96  * 32 * 32 * 4);
    float*    act2 = (float*)take((size_t)NBATCH * 192 * 16 * 16 * 4);
    float*    act3 = (float*)take((size_t)NBATCH * 384 * 16 * 16 * 4);
    float*    act4 = (float*)take((size_t)NBATCH * 768 * 16 * 16 * 4);
    _Float16* wh2  = (_Float16*)take((size_t)192 * 864  * 2);
    _Float16* wh3  = (_Float16*)take((size_t)384 * 1728 * 2);
    _Float16* wh4  = (_Float16*)take((size_t)768 * 3456 * 2);
    _Float16* wh5  = (_Float16*)take((size_t)768 * 6912 * 2);
    const int Cout_[5] = {96, 192, 384, 768, 768};
    float* sA[5]; float* sB[5];
    for (int l = 0; l < 5; ++l) {
        sA[l] = (float*)take((size_t)Cout_[l] * 4);
        sB[l] = (float*)take((size_t)Cout_[l] * 4);
    }
    int*   cnt = (int*)take((size_t)NBATCH * NCELL * NSEG * 4);
    float* den = (float*)take((size_t)NBATCH * NSEG * 4);

    // ---- prep: BN coefficients + f16 weights --------------------------------
    for (int l = 0; l < 5; ++l)
        bn_prep_kernel<<<(Cout_[l] + 63) / 64, 64, 0, stream>>>(
            Bb[l], G[l], Be[l], Mn[l], Vr[l], sA[l], sB[l], Cout_[l]);
    {
        const int n2 = 192 * 864, n3 = 384 * 1728, n4 = 768 * 3456, n5 = 768 * 6912;
        w2h_kernel<<<(n2 + 255) / 256, 256, 0, stream>>>(W[1], wh2, n2);
        w2h_kernel<<<(n3 + 255) / 256, 256, 0, stream>>>(W[2], wh3, n3);
        w2h_kernel<<<(n4 + 255) / 256, 256, 0, stream>>>(W[3], wh4, n4);
        w2h_kernel<<<(n5 + 255) / 256, 256, 0, stream>>>(W[4], wh5, n5);
    }

    // ---- conv stack ---------------------------------------------------------
    conv1_kernel<<<(NBATCH * 96 * 32 * 32) / 256, 256, 0, stream>>>(x, W[0], sA[0], sB[0], act1);

    // L2: 96->192, s2 p1, 32x32 -> 16x16   (N=8192 -> 128 tiles, M=192 -> 3)
    conv_wmma_kernel<<<dim3(128, 3),  256, 0, stream>>>(act1, wh2, sA[1], sB[1], nullptr, act2,
                                                        96, 32, 32, 192, 16, 16, 2, 1, 1);
    // L3: 192->384, s1 p1
    conv_wmma_kernel<<<dim3(128, 6),  256, 0, stream>>>(act2, wh3, sA[2], sB[2], nullptr, act3,
                                                        192, 16, 16, 384, 16, 16, 1, 1, 1);
    // L4: 384->768, s1 p1
    conv_wmma_kernel<<<dim3(128, 12), 256, 0, stream>>>(act3, wh4, sA[3], sB[3], nullptr, act4,
                                                        384, 16, 16, 768, 16, 16, 1, 1, 1);
    // L5: 768->768, s1 p0, 16x16 -> 14x14, no ReLU, + pos_embed, write x_convFeat
    //     (N = 32*196 = 6272 -> 98 tiles, M = 768 -> 12)
    conv_wmma_kernel<<<dim3(98, 12),  256, 0, stream>>>(act4, wh5, sA[4], sB[4], pos, cfeat,
                                                        768, 16, 16, 768, 14, 14, 1, 0, 0);

    // ---- segment pooling ----------------------------------------------------
    hipMemsetAsync(cnt, 0, (size_t)NBATCH * NCELL * NSEG * 4, stream);
    mask_kernel<<<(NBATCH * 56 * 56) / 256, 256, 0, stream>>>(seg, cnt);
    den_kernel<<<(NBATCH * NSEG + 255) / 256, 256, 0, stream>>>(cnt, den);
    feat_kernel<<<dim3(768 / 16, NBATCH), 256, 0, stream>>>(cfeat, cnt, den, xout);

    segcopy_kernel<<<(NBATCH * 64 * 64) / 256, 256, 0, stream>>>(seg, dout, NBATCH * 64 * 64);
}
